// s2b_attention_9199819948136
// MI455X (gfx1250) — matrix operand order repaired and checked by reading, not compiled, not run on hardware
//
#include <hip/hip_runtime.h>
#include <hip/hip_bf16.h>

typedef __attribute__((ext_vector_type(16))) _Float16 v16h;
typedef __attribute__((ext_vector_type(8)))  _Float16 v8h;
typedef __attribute__((ext_vector_type(8)))  float    v8f;
typedef __attribute__((ext_vector_type(4)))  float    v4f;

#define DIM   256
#define HEADS 8
#define HD    32
#define BATCH 4
#define NTOK  2304                 // 48*48
#define NROWS (BATCH * NTOK)       // 9216

// ---------------------------------------------------------------------------
// A-fragment loader for V_WMMA_F32_16X16X32_F16 (16x32 f16 A matrix).
// Lane (r = lane&15, hi = lane>>4) holds halfs[0..7]  = A[r][16*hi + i]
//                                    halfs[8..15] = A[r][16*hi + 8 + i]
// -> two contiguous 16B loads from a row-major A, p = &A[r][16*hi].
// ---------------------------------------------------------------------------
static __device__ __forceinline__ v16h load_a16(const _Float16* p) {
  v16h a;
  const v8h lo = *(const v8h*)(p);
  const v8h hp = *(const v8h*)(p + 8);
#pragma unroll
  for (int i = 0; i < 8; ++i) { a[i] = lo[i]; a[i + 8] = hp[i]; }
  return a;
}

static __device__ __forceinline__ v8f wmma16(v16h a, v16h b, v8f c) {
  // D = A(16x32) * B(32x16) + C, f32 accumulate
  return __builtin_amdgcn_wmma_f32_16x16x32_f16(false, a, false, b,
                                                (short)0, c, false, false);
}

// ---------------------------------------------------------------------------
// Kernel 0: weight transposes to f16 [out][in] + mask column vector.
// ---------------------------------------------------------------------------
__global__ void prep_kernel(const float* __restrict__ Wq, const float* __restrict__ Wk,
                            const float* __restrict__ Wv, const float* __restrict__ Wp,
                            const float* __restrict__ mask,
                            _Float16* __restrict__ Wtq, _Float16* __restrict__ Wtp,
                            float* __restrict__ mcol) {
  int idx = blockIdx.x * blockDim.x + threadIdx.x;
  const int NQKV = 768 * DIM;
  const int NP = DIM * DIM;
  if (idx < NQKV) {
    int j = idx / DIM, k = idx - j * DIM;   // j = output feature (0..767), k = input
    float v;
    if (j < 256)      v = Wq[k * DIM + j];
    else if (j < 512) v = Wk[k * DIM + (j - 256)];
    else              v = Wv[k * DIM + (j - 512)];
    Wtq[idx] = (_Float16)v;
  } else if (idx < NQKV + NP) {
    int t = idx - NQKV;
    int j = t / DIM, k = t - j * DIM;
    Wtp[t] = (_Float16)Wp[k * DIM + j];
  } else if (idx < NQKV + NP + NROWS) {
    int t = idx - NQKV - NP;
    mcol[t] = (mask[t] > 0.0f) ? 1.0f : 0.0f;   // mask is (B,1,48,48) == (B, N) flat
  }
}

// ---------------------------------------------------------------------------
// Kernel 1: LayerNorm, one wave (32 lanes) per row of 256, f16 output.
// ---------------------------------------------------------------------------
__global__ void ln_kernel(const float* __restrict__ x, const float* __restrict__ g,
                          const float* __restrict__ bta, _Float16* __restrict__ xn) {
  int row  = blockIdx.x * 8 + (threadIdx.x >> 5);
  int lane = threadIdx.x & 31;
  const float* xr = x + (size_t)row * DIM + lane * 8;
  v4f x0 = *(const v4f*)(xr);
  v4f x1 = *(const v4f*)(xr + 4);
  float s = 0.f;
#pragma unroll
  for (int i = 0; i < 4; ++i) s += x0[i] + x1[i];
  for (int m = 1; m < 32; m <<= 1) s += __shfl_xor(s, m, 32);
  float mu = s * (1.0f / DIM);
  float vq = 0.f;
#pragma unroll
  for (int i = 0; i < 4; ++i) {
    float d0 = x0[i] - mu, d1 = x1[i] - mu;
    vq += d0 * d0 + d1 * d1;
  }
  for (int m = 1; m < 32; m <<= 1) vq += __shfl_xor(vq, m, 32);
  float rs = rsqrtf(vq * (1.0f / DIM) + 1e-5f);
  const float* gp = g + lane * 8;
  const float* bp = bta + lane * 8;
  v8h o;
#pragma unroll
  for (int i = 0; i < 4; ++i) {
    o[i]     = (_Float16)((x0[i] - mu) * rs * gp[i]     + bp[i]);
    o[i + 4] = (_Float16)((x1[i] - mu) * rs * gp[i + 4] + bp[i + 4]);
  }
  *(v8h*)(xn + (size_t)row * DIM + lane * 8) = o;
}

// ---------------------------------------------------------------------------
// Kernel 2: QKV projection. 16 rows/block, 4 waves x 12 col-tiles, K=256.
// Wave index made explicitly uniform (readfirstlane -> scalar branches,
// EXEC stays full around WMMA). Tile-level double buffering: next tile's
// 8 B-fragments are loaded while the current 8-WMMA chain runs, so their
// live ranges force 16 b128 loads in flight under the WMMAs.
// Outputs: qh,kh natural [b][h][n][d] f16; vt transposed [b][h][d][n] f16.
// ---------------------------------------------------------------------------
__global__ void __launch_bounds__(128)
qkv_kernel(const _Float16* __restrict__ xn, const _Float16* __restrict__ Wt,
           const float* __restrict__ bq, const float* __restrict__ bk,
           const float* __restrict__ bv,
           _Float16* __restrict__ qh, _Float16* __restrict__ kh,
           _Float16* __restrict__ vt) {
  int row0 = blockIdx.x * 16;
  int b    = row0 / NTOK;
  int n0   = row0 - b * NTOK;
  int wv   = __builtin_amdgcn_readfirstlane(threadIdx.x >> 5);  // uniform wave id
  int lane = threadIdx.x & 31;
  int l15 = lane & 15, hi = lane >> 4;

  v16h a[8];
#pragma unroll
  for (int kc = 0; kc < 8; ++kc)
    a[kc] = load_a16(xn + (size_t)(row0 + l15) * DIM + kc * 32 + 16 * hi);

  const _Float16* wrow = Wt + (size_t)l15 * DIM + 16 * hi;  // + t*16*DIM per tile

  v16h bf[8];
#pragma unroll
  for (int kc = 0; kc < 8; ++kc)
    bf[kc] = *(const v16h*)(wrow + (size_t)wv * 16 * DIM + kc * 32);

  for (int t = wv; t < 48; t += 4) {
    v16h nbf[8];
    if (t + 4 < 48) {                      // scalar branch (t uniform)
#pragma unroll
      for (int kc = 0; kc < 8; ++kc)
        nbf[kc] = *(const v16h*)(wrow + (size_t)(t + 4) * 16 * DIM + kc * 32);
    } else {
#pragma unroll
      for (int kc = 0; kc < 8; ++kc) nbf[kc] = bf[kc];
    }

    v8f acc = {0.f, 0.f, 0.f, 0.f, 0.f, 0.f, 0.f, 0.f};
#pragma unroll
    for (int kc = 0; kc < 8; ++kc) acc = wmma16(a[kc], bf[kc], acc);

    int j0 = t * 16;
    if (j0 < 512) {                       // Q or K: natural [b][h][n][d]
      _Float16* dst = (j0 < 256) ? qh : kh;
      const float* bias = (j0 < 256) ? bq : bk;
      int jj = j0 & 255;
      int h = jj >> 5;
      int d = (jj & 31) + l15;
      float bc = bias[jj + l15];
      size_t base = (size_t)(b * HEADS + h) * NTOK * HD + d;
#pragma unroll
      for (int r = 0; r < 8; ++r) {
        int n = n0 + 8 * hi + r;
        dst[base + (size_t)n * HD] = (_Float16)(acc[r] + bc);
      }
    } else {                              // V: transposed [b][h][d][n]
      int jj = j0 - 512;
      int h = jj >> 5;
      int d = (jj & 31) + l15;
      float bc = bv[jj + l15];
      v8h ov;
#pragma unroll
      for (int r = 0; r < 8; ++r) ov[r] = (_Float16)(acc[r] + bc);
      *(v8h*)(vt + ((size_t)(b * HEADS + h) * HD + d) * NTOK + n0 + 8 * hi) = ov;
    }

#pragma unroll
    for (int kc = 0; kc < 8; ++kc) bf[kc] = nbf[kc];
  }
}

// ---------------------------------------------------------------------------
// Kernel 3: flash-style attention. One wave per (b,h,16-query tile).
// Software pipelined: chunk m+32's K/V fragments are loaded at the top of
// the iteration (overlapping chunk m's exp/LDS phase), and chunk m+64 is
// prefetched into cache (global_prefetch_b8).
// ---------------------------------------------------------------------------
__global__ void __launch_bounds__(32)
attn_kernel(const _Float16* __restrict__ qh, const _Float16* __restrict__ kh,
            const _Float16* __restrict__ vt, const float* __restrict__ mcol,
            _Float16* __restrict__ o) {
  __shared__ __align__(16) _Float16 Pld[16 * 40];   // 16 rows x 32 cols, stride 40
  int bid = blockIdx.x;
  int nt  = bid % (NTOK / 16);
  int bh  = bid / (NTOK / 16);
  int h = bh & 7, b = bh >> 3;
  int n0   = nt * 16;
  int lane = threadIdx.x & 31;
  int l15 = lane & 15, hi = lane >> 4;

  const _Float16* qb = qh + (size_t)(b * HEADS + h) * NTOK * HD;
  const _Float16* kp = kh + (size_t)(b * HEADS + h) * NTOK * HD
                          + (size_t)l15 * HD + 16 * hi;          // + m*HD per chunk
  const _Float16* vp0 = vt + (size_t)(b * HEADS + h) * HD * NTOK
                           + (size_t)l15 * NTOK + 16 * hi;       // + m per chunk
  const _Float16* vp1 = vp0 + (size_t)16 * NTOK;
  const float*    mb = mcol + (size_t)b * NTOK;

  v16h aq = load_a16(qb + (size_t)(n0 + l15) * HD + 16 * hi);

  const v8f vzero = {0.f, 0.f, 0.f, 0.f, 0.f, 0.f, 0.f, 0.f};
  v8f acc0 = vzero, acc1 = vzero;      // O accum, d-cols [0,16) and [16,32)
  float rsum[8];
#pragma unroll
  for (int r = 0; r < 8; ++r) rsum[r] = 0.f;

  // prologue: fragments for chunk 0
  v16h bk0 = *(const v16h*)(kp);
  v16h bk1 = *(const v16h*)(kp + 16 * HD);
  v16h bv0 = *(const v16h*)(vp0);
  v16h bv1 = *(const v16h*)(vp1);

  for (int m0 = 0; m0 < NTOK; m0 += 32) {
    int mn = m0 + 32;
    v16h nk0 = bk0, nk1 = bk1, nv0 = bv0, nv1 = bv1;
    if (mn < NTOK) {                     // issue next chunk's loads early
      nk0 = *(const v16h*)(kp + (size_t)mn * HD);
      nk1 = *(const v16h*)(kp + (size_t)(mn + 16) * HD);
      nv0 = *(const v16h*)(vp0 + mn);
      nv1 = *(const v16h*)(vp1 + mn);
      if (mn + 32 < NTOK) {              // cache prefetch for chunk m+64
        __builtin_prefetch(kp + (size_t)(mn + 32) * HD, 0, 1);
        __builtin_prefetch(vp0 + mn + 32, 0, 1);
        __builtin_prefetch(vp1 + mn + 32, 0, 1);
      }
    }

    v8f s0 = wmma16(aq, bk0, vzero);
    v8f s1 = wmma16(aq, bk1, vzero);

    float mc0 = mb[m0 + l15];
    float mc1 = mb[m0 + 16 + l15];
#pragma unroll
    for (int r = 0; r < 8; ++r) {
      float p0 = __expf(s0[r] * mc0);      // masked logit == 0 -> exp(0)=1 (ref semantics)
      float p1 = __expf(s1[r] * mc1);
      rsum[r] += p0 + p1;                  // this lane's key-columns only
      Pld[(8 * hi + r) * 40 + l15]      = (_Float16)p0;
      Pld[(8 * hi + r) * 40 + 16 + l15] = (_Float16)p1;
    }
    __syncthreads();                       // order LDS stores before A-frag loads

    v16h ap;
    {
      const v8h lo = *(const v8h*)(&Pld[l15 * 40 + 16 * hi]);
      const v8h hp = *(const v8h*)(&Pld[l15 * 40 + 16 * hi + 8]);
#pragma unroll
      for (int i = 0; i < 8; ++i) { ap[i] = lo[i]; ap[i + 8] = hp[i]; }
    }
    acc0 = wmma16(ap, bv0, acc0);
    acc1 = wmma16(ap, bv1, acc1);
    __syncthreads();                       // P tile consumed before next overwrite

    bk0 = nk0; bk1 = nk1; bv0 = nv0; bv1 = nv1;
  }

  // epilogue: reduce softmax denominator across the 16-lane half, normalize, store
#pragma unroll
  for (int r = 0; r < 8; ++r) {
    float tot = rsum[r];
    for (int m = 1; m < 16; m <<= 1) tot += __shfl_xor(tot, m, 32);
    float inv = 1.0f / tot;
    int n = n0 + 8 * hi + r;
    size_t ob = ((size_t)b * NTOK + n) * DIM + h * HD;
    o[ob + l15]      = (_Float16)(acc0[r] * inv);
    o[ob + 16 + l15] = (_Float16)(acc1[r] * inv);
  }
}

// ---------------------------------------------------------------------------
// Kernel 4: output projection O @ Wp + bp -> f32. Same double-buffered
// structure as qkv.
// ---------------------------------------------------------------------------
__global__ void __launch_bounds__(128)
proj_kernel(const _Float16* __restrict__ o, const _Float16* __restrict__ Wtp,
            const float* __restrict__ bp, float* __restrict__ out) {
  int row0 = blockIdx.x * 16;
  int wv   = __builtin_amdgcn_readfirstlane(threadIdx.x >> 5);  // uniform wave id
  int lane = threadIdx.x & 31;
  int l15 = lane & 15, hi = lane >> 4;

  v16h a[8];
#pragma unroll
  for (int kc = 0; kc < 8; ++kc)
    a[kc] = load_a16(o + (size_t)(row0 + l15) * DIM + kc * 32 + 16 * hi);

  const _Float16* wrow = Wtp + (size_t)l15 * DIM + 16 * hi;

  v16h bf[8];
#pragma unroll
  for (int kc = 0; kc < 8; ++kc)
    bf[kc] = *(const v16h*)(wrow + (size_t)wv * 16 * DIM + kc * 32);

  for (int t = wv; t < 16; t += 4) {
    v16h nbf[8];
    if (t + 4 < 16) {
#pragma unroll
      for (int kc = 0; kc < 8; ++kc)
        nbf[kc] = *(const v16h*)(wrow + (size_t)(t + 4) * 16 * DIM + kc * 32);
    } else {
#pragma unroll
      for (int kc = 0; kc < 8; ++kc) nbf[kc] = bf[kc];
    }

    v8f acc = {0.f, 0.f, 0.f, 0.f, 0.f, 0.f, 0.f, 0.f};
#pragma unroll
    for (int kc = 0; kc < 8; ++kc) acc = wmma16(a[kc], bf[kc], acc);

    int j0 = t * 16;
    float bc = bp[j0 + l15];
#pragma unroll
    for (int r = 0; r < 8; ++r)
      out[(size_t)(row0 + 8 * hi + r) * DIM + j0 + l15] = acc[r] + bc;

#pragma unroll
    for (int kc = 0; kc < 8; ++kc) bf[kc] = nbf[kc];
  }
}

// ---------------------------------------------------------------------------
extern "C" void kernel_launch(void* const* d_in, const int* in_sizes, int n_in,
                              void* d_out, int out_size, void* d_ws, size_t ws_size,
                              hipStream_t stream) {
  (void)in_sizes; (void)n_in; (void)out_size; (void)ws_size;
  const float* x    = (const float*)d_in[0];
  const float* mask = (const float*)d_in[1];
  const float* ln_g = (const float*)d_in[2];
  const float* ln_b = (const float*)d_in[3];
  const float* Wq   = (const float*)d_in[4];
  const float* bq   = (const float*)d_in[5];
  const float* Wk   = (const float*)d_in[6];
  const float* bk   = (const float*)d_in[7];
  const float* Wv   = (const float*)d_in[8];
  const float* bv   = (const float*)d_in[9];
  const float* Wp   = (const float*)d_in[10];
  const float* bp   = (const float*)d_in[11];
  float* out = (float*)d_out;

  char* ws = (char*)d_ws;
  const size_t SZH = (size_t)NROWS * DIM * sizeof(_Float16);   // 4.5 MB per tensor
  _Float16* xn  = (_Float16*)(ws);
  _Float16* qh  = (_Float16*)(ws + SZH);
  _Float16* kh  = (_Float16*)(ws + 2 * SZH);
  _Float16* vt  = (_Float16*)(ws + 3 * SZH);
  _Float16* oo  = xn;   // reuse: xn fully consumed by qkv before attention writes oo
  _Float16* Wtq = (_Float16*)(ws + 4 * SZH);
  _Float16* Wtp = (_Float16*)(ws + 4 * SZH + (size_t)768 * DIM * sizeof(_Float16));
  float*    mc  = (float*)(ws + 4 * SZH + (size_t)768 * DIM * sizeof(_Float16)
                              + (size_t)DIM * DIM * sizeof(_Float16));

  int prep_total = 768 * DIM + DIM * DIM + NROWS;
  prep_kernel<<<(prep_total + 255) / 256, 256, 0, stream>>>(Wq, Wk, Wv, Wp, mask,
                                                            Wtq, Wtp, mc);
  ln_kernel<<<NROWS / 8, 256, 0, stream>>>(x, ln_g, ln_b, xn);
  qkv_kernel<<<NROWS / 16, 128, 0, stream>>>(xn, Wtq, bq, bk, bv, qh, kh, vt);
  attn_kernel<<<BATCH * HEADS * (NTOK / 16), 32, 0, stream>>>(qh, kh, vt, mc, oo);
  proj_kernel<<<NROWS / 16, 128, 0, stream>>>(oo, Wtp, bp, out);
}
